// TransformerEncoder_38577396252869
// MI455X (gfx1250) — compile-verified, matrix operand order checked by reading, not run on hardware
//
#include <hip/hip_runtime.h>

// ---------------------------------------------------------------------------
// MI455X (gfx1250) ViT encoder. wave32. All matmuls via v_wmma_f32_16x16x32_bf16.
//   B=4 N=1024 C=768 D=12 H=12 HD=64 FF=3072, Mtok = B*N = 4096
// ~850 GFLOP forward => compute-bound (HBM time for bf16 weights ~7us at
// 23.3 TB/s). GEMMs: 64x64 register tiles per wave (16 WMMA : 16 b128 loads),
// weights L2-resident (192MB). Attention: flash-style, V tiles staged by the
// Tensor Data Mover (TENSORcnt-pipelined, double buffered, TDM pad feature
// used for LDS bank-conflict padding).
// ---------------------------------------------------------------------------

typedef __attribute__((ext_vector_type(16))) __bf16 v16bf;
typedef __attribute__((ext_vector_type(8)))  __bf16 v8bf;
typedef __attribute__((ext_vector_type(8)))  float  v8f;
typedef __attribute__((ext_vector_type(4)))  unsigned int u32x4;
typedef __attribute__((ext_vector_type(8)))  int i32x8;
typedef __attribute__((ext_vector_type(4)))  int i32x4;

#define MTOK 4096
#define CDIM 768
#define C3   2304
#define FFD  3072
#define NTOK 1024
#define NHEAD 12
#define DEPTH 12

#define VROW 66   // V tile LDS row stride in halves (64 + 2 pad from TDM)
#define PROW 40   // P tile LDS row stride in halves

static __device__ inline v8bf load8(const __bf16* p) { return *(const v8bf*)p; }

static __device__ inline v16bf make16(v8bf lo, v8bf hi) {
    v16bf r;
#pragma unroll
    for (int i = 0; i < 8; ++i) { r[i] = lo[i]; r[i + 8] = hi[i]; }
    return r;
}

static __device__ inline v8f wmma_bf16(v16bf a, v16bf b, v8f c) {
    // D = A(16x32 bf16) * B(32x16 bf16) + C(16x16 f32)
    return __builtin_amdgcn_wmma_f32_16x16x32_bf16(
        false, a, false, b, (short)0, c, false, false);
}

// TDM: stage one 32(keys) x 64(dims) bf16 tile, row stride 2304 elements,
// into LDS at byte offset ldsoff. pad_enable: +1 dword after every 32 dwords
// (= one 128B row) -> LDS row stride 66 halves. Tracked with TENSORcnt.
// This toolchain exposes the 6-arg builtin (extra int32x8, zero-filled).
static __device__ inline void tdm_load_vtile(const __bf16* gptr, unsigned ldsoff) {
    unsigned long long ga = (unsigned long long)gptr;
    u32x4 g0;
    g0[0] = 1u;                                        // count=1 (valid user D#)
    g0[1] = ldsoff;                                    // lds_addr (bytes)
    g0[2] = (unsigned)(ga & 0xffffffffu);              // global_addr[31:0]
    g0[3] = (unsigned)((ga >> 32) & 0x1ffffffu)        // global_addr[56:32]
          | (2u << 30);                                // type=2 ("image")
    i32x8 g1;
    g1[0] = (int)((1u << 16)                           // data_size=1 -> 2 bytes
                | (1u << 20)                           // pad_enable
                | (4u << 22)                           // pad_interval: 32 dwords
                | (0u << 25));                         // pad_amount: 1 dword
    g1[1] = (int)((64u & 0xffffu) << 16);              // tensor_dim0[15:0]=64
    g1[2] = (int)((32u & 0xffffu) << 16);              // dim0 hi=0 | tensor_dim1[15:0]=32
    g1[3] = (int)(64u << 16);                          // dim1 hi=0 | tile_dim0=64
    g1[4] = (int)(32u);                                // tile_dim1=32 | tile_dim2=0
    g1[5] = (int)(C3);                                 // tensor_dim0_stride=2304
    g1[6] = 0;                                         // stride hi | dim1_stride lo
    g1[7] = 0;
    i32x4 g2 = {0, 0, 0, 0};                           // 2-D tensor: groups 2/3 unused
    i32x4 g3 = {0, 0, 0, 0};
    i32x8 g4 = {0, 0, 0, 0, 0, 0, 0, 0};               // extra arg (6-arg toolchain)
    __builtin_amdgcn_tensor_load_to_lds(g0, g1, g2, g3, g4, 0);
}

// ---------------------------------------------------------------------------
// Elementwise helpers
// ---------------------------------------------------------------------------
__global__ __launch_bounds__(256) void k_add_pos(const float* __restrict__ a,
                                                 const float* __restrict__ b,
                                                 float* __restrict__ o) {
    size_t i = (size_t)blockIdx.x * 256 + threadIdx.x;
    o[i] = a[i] + b[i];
}

__global__ __launch_bounds__(256) void k_cvt_bf16(const float* __restrict__ in,
                                                  __bf16* __restrict__ out, int n) {
    int i = blockIdx.x * 256 + threadIdx.x;
    if (i < n) out[i] = (__bf16)in[i];
}

// ---------------------------------------------------------------------------
// LayerNorm over C=768; one row per block, 256 threads * 3 elems.
// ---------------------------------------------------------------------------
template <int OUTF32>
__global__ __launch_bounds__(256) void k_ln(const float* __restrict__ x,
                                            const float* __restrict__ g,
                                            const float* __restrict__ bt,
                                            void* __restrict__ outp) {
    __shared__ float s1[256], s2[256];
    const int tid = threadIdx.x;
    const float* xr = x + (size_t)blockIdx.x * CDIM;
    float a0 = xr[tid], a1 = xr[tid + 256], a2 = xr[tid + 512];
    s1[tid] = a0 + a1 + a2;
    s2[tid] = a0 * a0 + a1 * a1 + a2 * a2;
    __syncthreads();
#pragma unroll
    for (int st = 128; st > 0; st >>= 1) {
        if (tid < st) { s1[tid] += s1[tid + st]; s2[tid] += s2[tid + st]; }
        __syncthreads();
    }
    const float mu   = s1[0] * (1.0f / CDIM);
    const float var  = s2[0] * (1.0f / CDIM) - mu * mu;
    const float rinv = rsqrtf(var + 1e-5f);
    float av[3] = {a0, a1, a2};
#pragma unroll
    for (int k = 0; k < 3; ++k) {
        int c = tid + k * 256;
        float v = (av[k] - mu) * rinv * g[c] + bt[c];
        size_t idx = (size_t)blockIdx.x * CDIM + c;
        if (OUTF32) ((float*)outp)[idx] = v;
        else        ((__bf16*)outp)[idx] = (__bf16)v;
    }
}

// ---------------------------------------------------------------------------
// WMMA GEMM: out[M,N] = A[M,K](bf16) * W[N,K]^T(bf16) + bias[N]
// 8 waves as 2(M) x 4(N); wave tile 64x64 (4x4 WMMA subtiles) => block 128x256.
// 16 WMMA : 16 b128 loads per 32-wide K step (2x the reuse of a 32x64 tile).
// Fragment addressing per CDNA5 16-bit layouts: lane half -> K offset {0,8},
// vector halves 8..15 hold K+16..23 of the 32-wide step.
// ---------------------------------------------------------------------------
enum { EPI_BF16 = 0, EPI_GELU = 1, EPI_RESID = 2 };

template <int EPI>
__global__ __launch_bounds__(256) void k_gemm(const __bf16* __restrict__ A,
                                              const __bf16* __restrict__ W,
                                              const float* __restrict__ bias,
                                              void* __restrict__ outp,
                                              int K, int N) {
    const int lane = threadIdx.x & 31;
    const int wv   = threadIdx.x >> 5;
    const int wm   = wv & 1;
    const int wn   = wv >> 1;
    const int ln16 = lane & 15;
    const int off8 = (lane >> 4) * 8;
    const int m0 = blockIdx.x * 128 + wm * 64;
    const int n0 = blockIdx.y * 256 + wn * 64;

    const __bf16* pA[4];
    const __bf16* pW[4];
#pragma unroll
    for (int i = 0; i < 4; ++i) pA[i] = A + (size_t)(m0 + i * 16 + ln16) * K + off8;
#pragma unroll
    for (int j = 0; j < 4; ++j) pW[j] = W + (size_t)(n0 + j * 16 + ln16) * K + off8;

    v8f acc[4][4] = {};

    for (int kk = 0; kk < K; kk += 32) {
        v16bf a[4];
#pragma unroll
        for (int i = 0; i < 4; ++i)
            a[i] = make16(load8(pA[i] + kk), load8(pA[i] + kk + 16));
#pragma unroll
        for (int j = 0; j < 4; ++j) {
            v16bf bj = make16(load8(pW[j] + kk), load8(pW[j] + kk + 16));
            if (kk + 64 < K) __builtin_prefetch(pW[j] + kk + 64, 0, 1);  // global_prefetch_b8
#pragma unroll
            for (int i = 0; i < 4; ++i)
                acc[i][j] = wmma_bf16(a[i], bj, acc[i][j]);
        }
    }

#pragma unroll
    for (int j = 0; j < 4; ++j) {
        const int n = n0 + j * 16 + ln16;
        const float bj = bias[n];
#pragma unroll
        for (int i = 0; i < 4; ++i) {
#pragma unroll
            for (int r = 0; r < 8; ++r) {
                const int m = m0 + i * 16 + r + off8;  // D layout: row = r + 8*laneHalf
                const size_t idx = (size_t)m * N + n;
                float v = acc[i][j][r] + bj;
                if (EPI == EPI_BF16) {
                    ((__bf16*)outp)[idx] = (__bf16)v;
                } else if (EPI == EPI_GELU) {
                    v = 0.5f * v * (1.0f + erff(v * 0.70710678118654752f));
                    ((__bf16*)outp)[idx] = (__bf16)v;
                } else {  // residual accumulate, f32
                    float* op = (float*)outp;
                    op[idx] = op[idx] + v;
                }
            }
        }
    }
}

// ---------------------------------------------------------------------------
// Flash-style attention. grid(8, H, B), 256 thr = 8 independent waves.
// Wave handles 16 query rows of one (b,h). V tiles (32 keys x 64 dims) are
// staged by the Tensor Data Mover, double-buffered and pipelined on TENSORcnt.
// Per 32-key step: S = Q*K^T (4 WMMA), online softmax (shfl over 16-lane
// halves), P via LDS relayout to an A-fragment, O += P*V (4 WMMA).
// ---------------------------------------------------------------------------
__global__ __launch_bounds__(256) void k_attn(const __bf16* __restrict__ qkv,
                                              __bf16* __restrict__ obuf) {
    __shared__ __align__(16) __bf16 vstage[8][2][32 * VROW]; // per-wave double-buffered V
    __shared__ __align__(16) __bf16 pstage[8][16 * PROW];    // per-wave P tile

    const int lane = threadIdx.x & 31;
    const int wv   = threadIdx.x >> 5;
    const int ln16 = lane & 15;
    const int off8 = (lane >> 4) * 8;
    const int h = blockIdx.y;
    const int b = blockIdx.z;
    const int mrow = blockIdx.x * 128 + wv * 16;
    const float sc = 0.125f;                               // 1/sqrt(64)

    const __bf16* vbase = qkv + (size_t)b * NTOK * C3 + 2 * CDIM + h * 64;
    unsigned ldsv[2];
    ldsv[0] = (unsigned)(unsigned long long)&vstage[wv][0][0];
    ldsv[1] = (unsigned)(unsigned long long)&vstage[wv][1][0];

    // Q A-fragments for the two 32-wide head-dim chunks
    const __bf16* qp = qkv + ((size_t)(b * NTOK + mrow + ln16)) * C3 + h * 64 + off8;
    const v16bf qa0 = make16(load8(qp),      load8(qp + 16));
    const v16bf qa1 = make16(load8(qp + 32), load8(qp + 48));

    v8f   o[4] = {};
    float m_run[8], l_run[8];
#pragma unroll
    for (int r = 0; r < 8; ++r) { m_run[r] = -1e30f; l_run[r] = 0.0f; }

    // prime the TDM pipeline with the first V tile
    tdm_load_vtile(vbase, ldsv[0]);
    int buf = 0;

    for (int jj = 0; jj < NTOK; jj += 32) {
        // issue next V tile while we work on this one
        if (jj + 32 < NTOK) {
            tdm_load_vtile(vbase + (size_t)(jj + 32) * C3, ldsv[buf ^ 1]);
            __builtin_amdgcn_s_wait_tensorcnt(1);   // current tile done
        } else {
            __builtin_amdgcn_s_wait_tensorcnt(0);
        }
        asm volatile("" ::: "memory");

        // ---- S = Q*K^T for two 16-key tiles (head-dim contraction) ----
        const __bf16* kp0 = qkv + ((size_t)(b * NTOK + jj      + ln16)) * C3 + CDIM + h * 64 + off8;
        const __bf16* kp1 = qkv + ((size_t)(b * NTOK + jj + 16 + ln16)) * C3 + CDIM + h * 64 + off8;
        v8f s0 = {}, s1 = {};
        s0 = wmma_bf16(qa0, make16(load8(kp0),      load8(kp0 + 16)), s0);
        s0 = wmma_bf16(qa1, make16(load8(kp0 + 32), load8(kp0 + 48)), s0);
        s1 = wmma_bf16(qa0, make16(load8(kp1),      load8(kp1 + 16)), s1);
        s1 = wmma_bf16(qa1, make16(load8(kp1 + 32), load8(kp1 + 48)), s1);
        s0 = s0 * sc;
        s1 = s1 * sc;

        // ---- online softmax: row = r + 8*laneHalf, cols spread over 16 lanes ----
#pragma unroll
        for (int r = 0; r < 8; ++r) {
            float sm = fmaxf(s0[r], s1[r]);
#pragma unroll
            for (int mk = 1; mk <= 8; mk <<= 1) sm = fmaxf(sm, __shfl_xor(sm, mk, 32));
            const float nm = fmaxf(m_run[r], sm);
            const float corr = expf(m_run[r] - nm);
            const float p0 = expf(s0[r] - nm);
            const float p1 = expf(s1[r] - nm);
            float rs = p0 + p1;
#pragma unroll
            for (int mk = 1; mk <= 8; mk <<= 1) rs += __shfl_xor(rs, mk, 32);
            l_run[r] = l_run[r] * corr + rs;
            m_run[r] = nm;
            const int prow = r + off8;              // off8 == 8*laneHalf
            pstage[wv][prow * PROW + ln16]      = (__bf16)p0;
            pstage[wv][prow * PROW + 16 + ln16] = (__bf16)p1;
#pragma unroll
            for (int t = 0; t < 4; ++t) o[t][r] *= corr;
        }

        // ---- P A-fragment (16x32, K = 32 keys) from LDS ----
        const __bf16* pb = &pstage[wv][ln16 * PROW + off8];
        const v16bf pa = make16(load8(pb), load8(pb + 16));

        // ---- O += P * V : key-major B-fragment gather from TDM-staged V ----
        const __bf16* vs = &vstage[wv][buf][0];
#pragma unroll
        for (int t = 0; t < 4; ++t) {
            v16bf vb;
#pragma unroll
            for (int e = 0; e < 8; ++e) {
                vb[e]     = vs[(off8 + e)      * VROW + t * 16 + ln16];
                vb[e + 8] = vs[(off8 + e + 16) * VROW + t * 16 + ln16];
            }
            o[t] = wmma_bf16(pa, vb, o[t]);
        }
        buf ^= 1;
    }

    // ---- normalize and write O (cols = h*64 + d) ----
#pragma unroll
    for (int r = 0; r < 8; ++r) {
        const float dinv = 1.0f / l_run[r];
        const size_t row = (size_t)(b * NTOK + mrow + r + off8);
#pragma unroll
        for (int t = 0; t < 4; ++t)
            obuf[row * CDIM + h * 64 + t * 16 + ln16] = (__bf16)(o[t][r] * dinv);
    }
}

// ---------------------------------------------------------------------------
// Host launcher
// ---------------------------------------------------------------------------
extern "C" void kernel_launch(void* const* d_in, const int* in_sizes, int n_in,
                              void* d_out, int out_size, void* d_ws, size_t ws_size,
                              hipStream_t stream) {
    (void)in_sizes; (void)n_in; (void)out_size;
    const float* x      = (const float*)d_in[0];
    const float* pos    = (const float*)d_in[1];
    const float* qkv_w  = (const float*)d_in[2];
    const float* qkv_b  = (const float*)d_in[3];
    const float* proj_w = (const float*)d_in[4];
    const float* proj_b = (const float*)d_in[5];
    const float* fc1_w  = (const float*)d_in[6];
    const float* fc1_b  = (const float*)d_in[7];
    const float* fc2_w  = (const float*)d_in[8];
    const float* fc2_b  = (const float*)d_in[9];
    const float* ln1_g  = (const float*)d_in[10];
    const float* ln1_b  = (const float*)d_in[11];
    const float* ln2_g  = (const float*)d_in[12];
    const float* ln2_b  = (const float*)d_in[13];
    const float* nf_g   = (const float*)d_in[14];
    const float* nf_b   = (const float*)d_in[15];

    char* ws = (char*)d_ws;
    size_t off = 0;
    auto carve = [&](size_t bytes) -> char* {
        char* p = ws + off;
        off += (bytes + 255) & ~(size_t)255;
        return p;
    };
    float*  resid = (float*)  carve((size_t)MTOK * CDIM * 4);
    __bf16* hbuf  = (__bf16*) carve((size_t)MTOK * CDIM * 2);
    __bf16* qkvb  = (__bf16*) carve((size_t)MTOK * C3   * 2);
    __bf16* obuf  = (__bf16*) carve((size_t)MTOK * CDIM * 2);
    __bf16* ffb   = (__bf16*) carve((size_t)MTOK * FFD  * 2);
    __bf16* wqkv  = (__bf16*) carve((size_t)C3  * CDIM * 2);
    __bf16* wproj = (__bf16*) carve((size_t)CDIM * CDIM * 2);
    __bf16* wfc1  = (__bf16*) carve((size_t)FFD * CDIM * 2);
    __bf16* wfc2  = (__bf16*) carve((size_t)CDIM * FFD * 2);
    if (off > ws_size) return;

    // residual stream = x + pos
    k_add_pos<<<(MTOK * CDIM) / 256, 256, 0, stream>>>(x, pos, resid);

    for (int L = 0; L < DEPTH; ++L) {
        // per-layer weight conversion f32 -> bf16 (L2-resident afterwards)
        k_cvt_bf16<<<(C3 * CDIM + 255) / 256, 256, 0, stream>>>(qkv_w + (size_t)L * C3 * CDIM, wqkv, C3 * CDIM);
        k_cvt_bf16<<<(CDIM * CDIM + 255) / 256, 256, 0, stream>>>(proj_w + (size_t)L * CDIM * CDIM, wproj, CDIM * CDIM);
        k_cvt_bf16<<<(FFD * CDIM + 255) / 256, 256, 0, stream>>>(fc1_w + (size_t)L * FFD * CDIM, wfc1, FFD * CDIM);
        k_cvt_bf16<<<(CDIM * FFD + 255) / 256, 256, 0, stream>>>(fc2_w + (size_t)L * CDIM * FFD, wfc2, CDIM * FFD);

        // h = LN1(resid)
        k_ln<0><<<MTOK, 256, 0, stream>>>(resid, ln1_g + L * CDIM, ln1_b + L * CDIM, hbuf);
        // qkv = h @ Wqkv^T + b
        k_gemm<EPI_BF16><<<dim3(MTOK / 128, C3 / 256), 256, 0, stream>>>(hbuf, wqkv, qkv_b + (size_t)L * C3, qkvb, CDIM, C3);
        // attention
        k_attn<<<dim3(NTOK / 128, NHEAD, 4), 256, 0, stream>>>(qkvb, obuf);
        // resid += o @ Wproj^T + b
        k_gemm<EPI_RESID><<<dim3(MTOK / 128, CDIM / 256), 256, 0, stream>>>(obuf, wproj, proj_b + (size_t)L * CDIM, resid, CDIM, CDIM);
        // h = LN2(resid)
        k_ln<0><<<MTOK, 256, 0, stream>>>(resid, ln2_g + L * CDIM, ln2_b + L * CDIM, hbuf);
        // ff = gelu(h @ Wfc1^T + b)
        k_gemm<EPI_GELU><<<dim3(MTOK / 128, FFD / 256), 256, 0, stream>>>(hbuf, wfc1, fc1_b + (size_t)L * FFD, ffb, CDIM, FFD);
        // resid += ff @ Wfc2^T + b
        k_gemm<EPI_RESID><<<dim3(MTOK / 128, CDIM / 256), 256, 0, stream>>>(ffb, wfc2, fc2_b + (size_t)L * CDIM, resid, FFD, CDIM);
    }

    // final post-norm -> f32 output
    k_ln<1><<<MTOK, 256, 0, stream>>>(resid, nf_g, nf_b, (float*)d_out);
}
